// SparseLinearAttention_3238405342024
// MI455X (gfx1250) — matrix operand order repaired and checked by reading, not compiled, not run on hardware
//
#include <hip/hip_runtime.h>
#include <hip/hip_bf16.h>
#include <math.h>

// ---------------------------------------------------------------------------
// SparseLinearAttention for MI455X (gfx1250): bf16 WMMA (16x16x32) everywhere,
// f32 accumulate, flash-style online softmax for the block-sparse part.
// Stage-in uses packed bf16x2 LDS stores (32-bit DS ops).
// ---------------------------------------------------------------------------

typedef __attribute__((ext_vector_type(16))) __bf16 v16bf;
typedef __attribute__((ext_vector_type(2)))  __bf16 v2bf;
typedef __attribute__((ext_vector_type(8)))  float  v8f;

#define DEV __device__ __forceinline__

static constexpr int B_   = 2;
static constexpr int H_   = 16;
static constexpr int L_   = 2048;
static constexpr int D_   = 128;
static constexpr int NBLK = 32;   // L/64 (BLKQ == BLKK == 64)
static constexpr int TOPK = 4;    // max(1, int(0.125*32))
static constexpr float SCALE = 0.08838834764831845f; // 1/sqrt(128)

// --- WMMA fragment loaders (ISA 7.12.2 layouts, wave32) --------------------
// A (16-bit, 16x32): lane holds row M=lane&15; elems K = k0 + 16*(v/4) + 8*hi
//                    + 2*(v%4) + h  -> two contiguous 8-elem (16B) chunks.
DEV v16bf load_a_frag(const __bf16* base, int stride, int m, int k0, int hi) {
  const __bf16* p = base + m * stride + k0 + 8 * hi;
  v16bf f;
#pragma unroll
  for (int e = 0; e < 8; ++e) f[e] = p[e];
#pragma unroll
  for (int e = 0; e < 8; ++e) f[e + 8] = p[e + 16];
  return f;
}

// B (16-bit, 32x16): lane holds col N=lane&15; elems K = k0 + 8*hi + e (e<8),
//                    k0 + 16 + 8*hi + (e-8) (e>=8) -> two 8-elem chunks.
DEV v16bf load_b_frag(const __bf16* base, int stride, int n, int k0, int hi) {
  const __bf16* p = base + n * stride + k0 + 8 * hi;
  v16bf f;
#pragma unroll
  for (int e = 0; e < 8; ++e) f[e] = p[e];
#pragma unroll
  for (int e = 0; e < 8; ++e) f[e + 8] = p[e + 16];
  return f;
}

DEV v8f wmma_bf16(v16bf a, v16bf b, v8f c) {
  // (neg_a, A, neg_b, B, c_mod, C, reuse_a, reuse_b)
  return __builtin_amdgcn_wmma_f32_16x16x32_bf16(false, a, false, b,
                                                 (short)0, c, false, false);
}

DEV void store_pk(__bf16* dst, float a, float b) {  // one 32-bit DS store
  v2bf pk; pk[0] = (__bf16)a; pk[1] = (__bf16)b;
  *(v2bf*)dst = pk;
}

// ---------------------------------------------------------------------------
// K1: per-block means over 64 rows of D columns (for q and k)
// grid = B*H*NBLK, block = 128
__global__ void block_mean_kernel(const float* __restrict__ x,
                                  float* __restrict__ xb) {
  const int blk = blockIdx.x;
  const int d   = threadIdx.x;           // 0..127
  const float* p = x + (size_t)blk * 64 * D_ + d;
  float s = 0.f;
#pragma unroll
  for (int r = 0; r < 64; ++r) s += p[(size_t)r * D_];
  xb[(size_t)blk * D_ + d] = s * (1.0f / 64.0f);
}

// ---------------------------------------------------------------------------
// K2: 32x32 block scores + top-4 per row. grid = B*H, block = 32 (one wave).
__global__ void topk_kernel(const float* __restrict__ qb,
                            const float* __restrict__ kb,
                            int* __restrict__ lut) {
  __shared__ float qs[NBLK * D_];
  __shared__ float ks[NBLK * D_];
  const int bh = blockIdx.x;
  for (int i = threadIdx.x; i < NBLK * D_; i += 32) {
    qs[i] = qb[(size_t)bh * NBLK * D_ + i];
    ks[i] = kb[(size_t)bh * NBLK * D_ + i];
  }
  __syncthreads();
  const int row = threadIdx.x;           // q-block id
  float sc[NBLK];
  for (int kk = 0; kk < NBLK; ++kk) {
    float s = 0.f;
    for (int d = 0; d < D_; ++d) s += qs[row * D_ + d] * ks[kk * D_ + d];
    sc[kk] = s;
  }
  for (int t = 0; t < TOPK; ++t) {       // argmax, first-wins (matches top_k)
    float best = -INFINITY; int bi = 0;
    for (int kk = 0; kk < NBLK; ++kk)
      if (sc[kk] > best) { best = sc[kk]; bi = kk; }
    lut[((size_t)bh * NBLK + row) * TOPK + t] = bi;
    sc[bi] = -INFINITY;
  }
}

// ---------------------------------------------------------------------------
// K3: block-sparse attention with online softmax.
// grid = B*H*NBLK, block = 128 (4 waves; wave w owns q-rows [16w,16w+16)).
__global__ void __launch_bounds__(128)
sparse_attn_kernel(const float* __restrict__ q, const float* __restrict__ k,
                   const float* __restrict__ v, const int* __restrict__ lut,
                   float* __restrict__ out) {
  __shared__ __align__(32) __bf16 q_lds[64 * D_];    // [m][d]
  __shared__ __align__(32) __bf16 k_lds[64 * D_];    // [j][d]
  __shared__ __align__(32) __bf16 vT_lds[D_ * 80];   // [e][j], padded stride
  __shared__ __align__(32) __bf16 p_lds[64 * 64];    // [m][j]

  const int gid  = blockIdx.x;
  const int qblk = gid % NBLK;
  const int bh   = gid / NBLK;
  const int tid  = threadIdx.x;
  const int wave = tid >> 5;
  const int lane = tid & 31;
  const int lo = lane & 15, hi = lane >> 4;
  const int m0 = wave * 16;
  const size_t head_off = (size_t)bh * L_ * D_;

  { // stage Q tile (64x128) as bf16, packed pair stores
    const float2* qp2 = (const float2*)(q + head_off + (size_t)qblk * 64 * D_);
    for (int i = tid; i < 64 * D_ / 2; i += 128) {
      const float2 x = qp2[i];
      store_pk(q_lds + 2 * i, x.x, x.y);
    }
  }
  __syncthreads();

  v16bf aq[4];
#pragma unroll
  for (int kc = 0; kc < 4; ++kc)
    aq[kc] = load_a_frag(q_lds, D_, m0 + lo, kc * 32, hi);

  const v8f vzero = {};
  v8f acc_o[8];
#pragma unroll
  for (int nt = 0; nt < 8; ++nt) acc_o[nt] = vzero;
  float m_run[8], l_run[8];
#pragma unroll
  for (int r = 0; r < 8; ++r) { m_run[r] = -INFINITY; l_run[r] = 0.f; }

  for (int t = 0; t < TOPK; ++t) {
    __syncthreads();  // previous iteration's K/V reads complete
    const int blkid = lut[((size_t)bh * NBLK + qblk) * TOPK + t];
    const float* kp = k + head_off + (size_t)blkid * 64 * D_;
    const float* vp = v + head_off + (size_t)blkid * 64 * D_;
    { // K row-major: pack along d
      const float2* kp2 = (const float2*)kp;
      for (int i = tid; i < 64 * D_ / 2; i += 128) {
        const float2 x = kp2[i];
        store_pk(k_lds + 2 * i, x.x, x.y);
      }
      // V transposed: pack along r (conflict-free b32 stores)
      for (int i = tid; i < 64 * D_ / 2; i += 128) {
        const int pr = i & 31, d = i >> 5, r = pr * 2;
        store_pk(vT_lds + d * 80 + r, vp[(size_t)r * D_ + d],
                 vp[(size_t)(r + 1) * D_ + d]);
      }
    }
    __syncthreads();

    // S = Q·K^T * scale : 4 column tiles of 16
    v8f s[4];
#pragma unroll
    for (int nt = 0; nt < 4; ++nt) {
      v8f acc = vzero;
#pragma unroll
      for (int kc = 0; kc < 4; ++kc) {
        v16bf bk = load_b_frag(k_lds, D_, nt * 16 + lo, kc * 32, hi);
        acc = wmma_bf16(aq[kc], bk, acc);
      }
      s[nt] = acc * SCALE;
    }

    // online softmax (row = r + 8*hi; reduce over the 16-lane half-wave)
#pragma unroll
    for (int r = 0; r < 8; ++r) {
      float mx = s[0][r];
#pragma unroll
      for (int nt = 1; nt < 4; ++nt) mx = fmaxf(mx, s[nt][r]);
#pragma unroll
      for (int msk = 1; msk < 16; msk <<= 1)
        mx = fmaxf(mx, __shfl_xor(mx, msk, 32));
      const float mnew = fmaxf(m_run[r], mx);
      const float corr = __expf(m_run[r] - mnew);
      float lsum = 0.f;
#pragma unroll
      for (int nt = 0; nt < 4; ++nt) {
        const float p = __expf(s[nt][r] - mnew);
        s[nt][r] = p;
        lsum += p;
      }
#pragma unroll
      for (int msk = 1; msk < 16; msk <<= 1)
        lsum += __shfl_xor(lsum, msk, 32);
      l_run[r] = l_run[r] * corr + lsum;
      m_run[r] = mnew;
#pragma unroll
      for (int nt = 0; nt < 8; ++nt) acc_o[nt][r] *= corr;
      const int prow = m0 + r + 8 * hi;           // wave-local rows only
#pragma unroll
      for (int nt = 0; nt < 4; ++nt)
        p_lds[prow * 64 + nt * 16 + lo] = (__bf16)s[nt][r];
    }
    __builtin_amdgcn_wave_barrier();  // P in LDS is wave-local; DS is in-order

    // O += P·V
    v16bf ap0 = load_a_frag(p_lds, 64, m0 + lo, 0, hi);
    v16bf ap1 = load_a_frag(p_lds, 64, m0 + lo, 32, hi);
#pragma unroll
    for (int nt = 0; nt < 8; ++nt) {
      v16bf bv0 = load_b_frag(vT_lds, 80, nt * 16 + lo, 0, hi);
      v16bf bv1 = load_b_frag(vT_lds, 80, nt * 16 + lo, 32, hi);
      acc_o[nt] = wmma_bf16(ap0, bv0, acc_o[nt]);
      acc_o[nt] = wmma_bf16(ap1, bv1, acc_o[nt]);
    }
  }

  // o_s = O / l  (overwrite d_out; linear part adds on top later)
#pragma unroll
  for (int r = 0; r < 8; ++r) {
    const float inv = 1.0f / l_run[r];
    const size_t row = head_off + ((size_t)qblk * 64 + m0 + r + 8 * hi) * D_;
#pragma unroll
    for (int nt = 0; nt < 8; ++nt)
      out[row + nt * 16 + lo] = acc_o[nt][r] * inv;
  }
}

// ---------------------------------------------------------------------------
// K4: kvsum = phi_k^T @ V (128x128 per head), ksum = sum_l phi_k.
// grid = B*H, block = 128. Accumulate fully in registers (no zeroed scratch).
__global__ void __launch_bounds__(128)
kvsum_kernel(const float* __restrict__ k, const float* __restrict__ v,
             float* __restrict__ kvsum, float* __restrict__ ksum) {
  __shared__ __align__(32) __bf16 phiT[D_ * 80];  // [d][l-chunk], stride 80
  __shared__ __align__(32) __bf16 vT[D_ * 80];    // [e][l-chunk]

  const int bh  = blockIdx.x;
  const int tid = threadIdx.x;
  const int wave = tid >> 5, lane = tid & 31;
  const int lo = lane & 15, hi = lane >> 4;
  const size_t head_off = (size_t)bh * L_ * D_;

  const v8f vzero = {};
  v8f acc[2][8];
#pragma unroll
  for (int j = 0; j < 2; ++j)
#pragma unroll
    for (int nt = 0; nt < 8; ++nt) acc[j][nt] = vzero;
  float ks_acc = 0.f;  // thread tid owns d = tid

  for (int c = 0; c < L_ / 64; ++c) {
    __syncthreads();
    if (tid < 64) {  // softmax over D for row l = c*64 + tid
      const float* kr = k + head_off + (size_t)(c * 64 + tid) * D_;
      float mx = -INFINITY;
      for (int d = 0; d < D_; ++d) mx = fmaxf(mx, kr[d]);
      float sum = 0.f;
      for (int d = 0; d < D_; ++d) sum += __expf(kr[d] - mx);
      const float rinv = 1.0f / sum;
      for (int d = 0; d < D_; ++d)
        phiT[d * 80 + tid] = (__bf16)(__expf(kr[d] - mx) * rinv);
    }
    { // V transposed chunk: packed pair stores along r
      const float* vp = v + head_off + (size_t)c * 64 * D_;
      for (int i = tid; i < 64 * D_ / 2; i += 128) {
        const int pr = i & 31, d = i >> 5, r = pr * 2;
        store_pk(vT + d * 80 + r, vp[(size_t)r * D_ + d],
                 vp[(size_t)(r + 1) * D_ + d]);
      }
    }
    __syncthreads();

    { // ksum partial for d = tid
      float ss = 0.f;
      for (int l = 0; l < 64; ++l) ss += (float)phiT[tid * 80 + l];
      ks_acc += ss;
    }
    // kvsum tiles: wave w owns tile-rows {2w, 2w+1} of the 8x8 tile grid
#pragma unroll
    for (int kc = 0; kc < 2; ++kc) {
      v16bf a0 = load_a_frag(phiT, 80, (2 * wave + 0) * 16 + lo, kc * 32, hi);
      v16bf a1 = load_a_frag(phiT, 80, (2 * wave + 1) * 16 + lo, kc * 32, hi);
#pragma unroll
      for (int nt = 0; nt < 8; ++nt) {
        v16bf b = load_b_frag(vT, 80, nt * 16 + lo, kc * 32, hi);
        acc[0][nt] = wmma_bf16(a0, b, acc[0][nt]);
        acc[1][nt] = wmma_bf16(a1, b, acc[1][nt]);
      }
    }
  }

#pragma unroll
  for (int j = 0; j < 2; ++j)
#pragma unroll
    for (int r = 0; r < 8; ++r) {
      const int drow = (2 * wave + j) * 16 + r + 8 * hi;
#pragma unroll
      for (int nt = 0; nt < 8; ++nt)
        kvsum[(size_t)bh * D_ * D_ + (size_t)drow * D_ + nt * 16 + lo] =
            acc[j][nt][r];
    }
  ksum[(size_t)bh * D_ + tid] = ks_acc;
}

// ---------------------------------------------------------------------------
// K5: num = phi_q @ kvsum; o_l = num/den; out += o_l @ w_proj^T + b_proj.
// grid = B*H*NBLK (64-row strips), block = 128.
__global__ void __launch_bounds__(128)
linear_out_kernel(const float* __restrict__ q, const float* __restrict__ kvsum,
                  const float* __restrict__ ksum,
                  const float* __restrict__ w_proj,
                  const float* __restrict__ b_proj, float* __restrict__ out) {
  __shared__ __align__(32) __bf16 phiq[64 * D_];   // [l][d]; reused for o_l
  __shared__ __align__(32) __bf16 buf[D_ * 144];   // kvsum^T, then w_proj
  __shared__ float ksum_lds[D_];
  __shared__ float den_lds[64];

  const int gid = blockIdx.x;
  const int lb  = gid % NBLK;
  const int bh  = gid / NBLK;
  const int tid = threadIdx.x;
  const int wave = tid >> 5, lane = tid & 31;
  const int lo = lane & 15, hi = lane >> 4;
  const int m0 = wave * 16;
  const size_t head_off = (size_t)bh * L_ * D_;

  if (tid < D_) ksum_lds[tid] = ksum[(size_t)bh * D_ + tid];
  { // kvsum^T: buf[e][d], packed along d
    const float* kv = kvsum + (size_t)bh * D_ * D_;
    for (int i = tid; i < D_ * D_ / 2; i += 128) {
      const int e = i & (D_ - 1), d = (i >> 7) * 2;
      store_pk(buf + e * 144 + d, kv[(size_t)d * D_ + e],
               kv[(size_t)(d + 1) * D_ + e]);
    }
  }
  __syncthreads();

  if (tid < 64) {  // phi_q row + den
    const float* qr = q + head_off + (size_t)(lb * 64 + tid) * D_;
    float mx = -INFINITY;
    for (int d = 0; d < D_; ++d) mx = fmaxf(mx, qr[d]);
    float sum = 0.f;
    for (int d = 0; d < D_; ++d) sum += __expf(qr[d] - mx);
    const float rinv = 1.0f / sum;
    float den = 1e-5f;
    for (int d = 0; d < D_; ++d) {
      const float p = __expf(qr[d] - mx) * rinv;
      phiq[tid * D_ + d] = (__bf16)p;
      den += p * ksum_lds[d];
    }
    den_lds[tid] = den;
  }
  __syncthreads();

  const v8f vzero = {};
  v8f acc[8];
#pragma unroll
  for (int nt = 0; nt < 8; ++nt) acc[nt] = vzero;
#pragma unroll
  for (int kc = 0; kc < 4; ++kc) {  // num = phi_q @ kvsum
    v16bf a = load_a_frag(phiq, D_, m0 + lo, kc * 32, hi);
#pragma unroll
    for (int nt = 0; nt < 8; ++nt) {
      v16bf b = load_b_frag(buf, 144, nt * 16 + lo, kc * 32, hi);
      acc[nt] = wmma_bf16(a, b, acc[nt]);
    }
  }
  __builtin_amdgcn_wave_barrier();
  // o_l = num/den staged back into phiq (wave-local rows only)
#pragma unroll
  for (int r = 0; r < 8; ++r) {
    const int row = m0 + r + 8 * hi;
    const float rinv = 1.0f / den_lds[row];
#pragma unroll
    for (int nt = 0; nt < 8; ++nt)
      phiq[row * D_ + nt * 16 + lo] = (__bf16)(acc[nt][r] * rinv);
  }
  __syncthreads();
  { // buf <- w_proj (row-major), packed float2 loads + b32 stores
    const float2* w2 = (const float2*)w_proj;
    for (int i = tid; i < D_ * D_ / 2; i += 128) {
      const int e = i >> 6, d = (i & 63) * 2;
      const float2 x = w2[i];
      store_pk(buf + e * 144 + d, x.x, x.y);
    }
  }
  __syncthreads();

  v8f acc2[8];
#pragma unroll
  for (int nt = 0; nt < 8; ++nt) acc2[nt] = vzero;
#pragma unroll
  for (int kc = 0; kc < 4; ++kc) {  // o_l @ w_proj^T
    v16bf a = load_a_frag(phiq, D_, m0 + lo, kc * 32, hi);
#pragma unroll
    for (int nt = 0; nt < 8; ++nt) {
      v16bf b = load_b_frag(buf, 144, nt * 16 + lo, kc * 32, hi);
      acc2[nt] = wmma_bf16(a, b, acc2[nt]);
    }
  }
#pragma unroll
  for (int r = 0; r < 8; ++r) {
    const size_t row = head_off + ((size_t)lb * 64 + m0 + r + 8 * hi) * D_;
#pragma unroll
    for (int nt = 0; nt < 8; ++nt) {
      const int e = nt * 16 + lo;
      out[row + e] += acc2[nt][r] + b_proj[e];
    }
  }
}

// ---------------------------------------------------------------------------
extern "C" void kernel_launch(void* const* d_in, const int* in_sizes, int n_in,
                              void* d_out, int out_size, void* d_ws,
                              size_t ws_size, hipStream_t stream) {
  const float* qf = (const float*)d_in[0];
  const float* kf = (const float*)d_in[1];
  const float* vf = (const float*)d_in[2];
  const float* wf = (const float*)d_in[3];
  const float* bf = (const float*)d_in[4];
  float* out = (float*)d_out;

  // workspace layout (floats): qb | kb | kvsum | ksum | lut(int)
  float* ws  = (float*)d_ws;
  float* qb  = ws;                       // B*H*32*128 = 131072
  float* kb  = ws + 131072;              // 131072
  float* kvs = ws + 262144;              // B*H*128*128 = 524288
  float* ksm = ws + 786432;              // B*H*128 = 4096
  int*   lut = (int*)(ws + 790528);      // B*H*32*4 = 4096 ints

  block_mean_kernel<<<B_ * H_ * NBLK, 128, 0, stream>>>(qf, qb);
  block_mean_kernel<<<B_ * H_ * NBLK, 128, 0, stream>>>(kf, kb);
  topk_kernel<<<B_ * H_, 32, 0, stream>>>(qb, kb, lut);
  sparse_attn_kernel<<<B_ * H_ * NBLK, 128, 0, stream>>>(qf, kf, vf, lut, out);
  kvsum_kernel<<<B_ * H_, 128, 0, stream>>>(kf, vf, kvs, ksm);
  linear_out_kernel<<<B_ * H_ * NBLK, 128, 0, stream>>>(qf, kvs, ksm, wf, bf,
                                                        out);
}
